// SparseUNet_17136919511612
// MI455X (gfx1250) — compile-verified, matrix operand order checked by reading, not run on hardware
//
#include <hip/hip_runtime.h>

// ---------------------------------------------------------------------------
// CDNA5 (gfx1250) implicit-GEMM 3D U-Net, bf16 WMMA with f32 accumulation.
// Activations: channels-last bf16 (stored as ushort bit patterns).
// Weights: pre-packed on device into V_WMMA_F32_16X16X32_BF16 B-fragment order
//          with cout-tile innermost so 4 B blocks per (tap,kchunk) are
//          contiguous. Register blocking: 32 voxels x 64 couts per wave
//          (2 voxel subtiles x 4 cout tiles = 8 accumulators; A reused 4x,
//          B reused 2x -> ~1 WMMA per 1.5 b128 loads).
// ---------------------------------------------------------------------------

typedef __attribute__((ext_vector_type(16))) __bf16       v16bf;
typedef __attribute__((ext_vector_type(8)))  float        v8f;
typedef __attribute__((ext_vector_type(4)))  unsigned int u32x4;

union Frag { v16bf v; u32x4 q[2]; };

__device__ __forceinline__ unsigned short f2bf(float f) {
  unsigned int u = __float_as_uint(f);
  u += 0x7FFFu + ((u >> 16) & 1u);           // round-to-nearest-even
  return (unsigned short)(u >> 16);
}
__device__ __forceinline__ float bf2f(unsigned short h) {
  return __uint_as_float(((unsigned int)h) << 16);
}

// ---------------------------------------------------------------------------
// Implicit-GEMM conv. One wave computes a 32-voxel x 64-cout tile.
//   MODE 0: k3, stride 1, pad 1                      (Sout == Sin)
//   MODE 1: k3, stride 2, pad 1                      (Sout == Sin/2)
//   MODE 2: transpose k3 (lhs_dilation 2, pad(1,2))  (Sout == 2*Sin)
//   MODE 3: 1x1                                      (Sout == Sin)
// KC = Cin/32 (compile-time).
// A fragment (16x32 bf16, ISA 7.12.2): lanes 0-15 hold row M=lane with
// K = {c0..c0+7, c0+16..c0+23}; lanes 16-31 hold K = {c0+8..15, c0+24..31}.
// Out-of-bounds taps read a zeroed scratch row (branchless address select).
// D (16x16 f32): lane holds column N=lane&15, rows r(+8 for upper half-wave).
// ---------------------------------------------------------------------------
template <int MODE, int KC>
__global__ __launch_bounds__(32) void conv3d_wmma(
    const unsigned short* __restrict__ act,
    const unsigned short* __restrict__ wpk,
    const unsigned short* __restrict__ zrow,
    float* __restrict__ out,
    int Sin, int Sout, int Cout)
{
  const int lane = threadIdx.x;
  const int m    = lane & 15;
  const int hi   = lane >> 4;

  // two voxel subtiles per wave
  const int v0 = blockIdx.x * 32 + m;
  const int v1 = v0 + 16;
  const int HW = Sout * Sout;
  const int od0 = v0 / HW, q0 = v0 - od0 * HW, oh0 = q0 / Sout, ow0 = q0 - oh0 * Sout;
  const int od1 = v1 / HW, q1 = v1 - od1 * HW, oh1 = q1 / Sout, ow1 = q1 - oh1 * Sout;

  constexpr int T   = (MODE == 3) ? 1 : 27;
  constexpr int Cin = KC * 32;
  const int NCT  = Cout >> 4;           // total cout tiles in packed layout
  const int cA0  = hi ? 8  : 0;         // channel offset, frag elems 0-7
  const int cA1  = hi ? 24 : 16;        // channel offset, frag elems 8-15

  const unsigned short* wbase =
      wpk + (size_t)(blockIdx.y * 4) * 512 + lane * 16;

  v8f acc[2][4] = {};

  for (int t = 0; t < T; ++t) {
    int kd = 0, kh = 0, kw = 0;
    if (MODE != 3) {
      kd = t / 9; const int rr = t - kd * 9; kh = rr / 3; kw = rr - kh * 3;
    }

    const unsigned short* ap0;
    const unsigned short* ap1;
    if (MODE == 0 || MODE == 1) {
      const int s = (MODE == 1) ? 2 : 1;
      const int id0 = od0 * s - 1 + kd, ih0 = oh0 * s - 1 + kh, iw0 = ow0 * s - 1 + kw;
      const int id1 = od1 * s - 1 + kd, ih1 = oh1 * s - 1 + kh, iw1 = ow1 * s - 1 + kw;
      const bool ok0 = (unsigned)id0 < (unsigned)Sin && (unsigned)ih0 < (unsigned)Sin &&
                       (unsigned)iw0 < (unsigned)Sin;
      const bool ok1 = (unsigned)id1 < (unsigned)Sin && (unsigned)ih1 < (unsigned)Sin &&
                       (unsigned)iw1 < (unsigned)Sin;
      ap0 = ok0 ? act + ((size_t)(id0 * Sin + ih0) * Sin + iw0) * Cin : zrow;
      ap1 = ok1 ? act + ((size_t)(id1 * Sin + ih1) * Sin + iw1) * Cin : zrow;
    } else if (MODE == 2) {
      // dilated-input gather: t_in = o-1+k must be even, t_in/2 in range
      const int td0 = od0 - 1 + kd, th0 = oh0 - 1 + kh, tw0 = ow0 - 1 + kw;
      const int td1 = od1 - 1 + kd, th1 = oh1 - 1 + kh, tw1 = ow1 - 1 + kw;
      const int id0 = td0 >> 1, ih0 = th0 >> 1, iw0 = tw0 >> 1;
      const int id1 = td1 >> 1, ih1 = th1 >> 1, iw1 = tw1 >> 1;
      const bool ok0 = (((td0 | th0 | tw0) & 1) == 0) &&
                       (unsigned)id0 < (unsigned)Sin && (unsigned)ih0 < (unsigned)Sin &&
                       (unsigned)iw0 < (unsigned)Sin;
      const bool ok1 = (((td1 | th1 | tw1) & 1) == 0) &&
                       (unsigned)id1 < (unsigned)Sin && (unsigned)ih1 < (unsigned)Sin &&
                       (unsigned)iw1 < (unsigned)Sin;
      ap0 = ok0 ? act + ((size_t)(id0 * Sin + ih0) * Sin + iw0) * Cin : zrow;
      ap1 = ok1 ? act + ((size_t)(id1 * Sin + ih1) * Sin + iw1) * Cin : zrow;
    } else {
      ap0 = act + ((size_t)(od0 * Sin + oh0) * Sin + ow0) * Cin;
      ap1 = act + ((size_t)(od1 * Sin + oh1) * Sin + ow1) * Cin;
    }

    const unsigned short* wt = wbase + (size_t)(t * KC) * NCT * 512;
    // prefetch next tap's weight blocks (global_prefetch_b8)
    __builtin_prefetch(wt + (size_t)KC * NCT * 512, 0, 3);

#pragma unroll
    for (int kc = 0; kc < KC; ++kc) {
      Frag a0, a1, b0, b1, b2, b3;
      a0.q[0] = *(const u32x4*)(ap0 + kc * 32 + cA0);
      a0.q[1] = *(const u32x4*)(ap0 + kc * 32 + cA1);
      a1.q[0] = *(const u32x4*)(ap1 + kc * 32 + cA0);
      a1.q[1] = *(const u32x4*)(ap1 + kc * 32 + cA1);
      const unsigned short* wp = wt + (size_t)kc * NCT * 512;
      b0.q[0] = *(const u32x4*)(wp);        b0.q[1] = *(const u32x4*)(wp + 8);
      b1.q[0] = *(const u32x4*)(wp + 512);  b1.q[1] = *(const u32x4*)(wp + 520);
      b2.q[0] = *(const u32x4*)(wp + 1024); b2.q[1] = *(const u32x4*)(wp + 1032);
      b3.q[0] = *(const u32x4*)(wp + 1536); b3.q[1] = *(const u32x4*)(wp + 1544);
      acc[0][0] = __builtin_amdgcn_wmma_f32_16x16x32_bf16(
          false, a0.v, false, b0.v, (short)0, acc[0][0], false, false);
      acc[1][0] = __builtin_amdgcn_wmma_f32_16x16x32_bf16(
          false, a1.v, false, b0.v, (short)0, acc[1][0], false, false);
      acc[0][1] = __builtin_amdgcn_wmma_f32_16x16x32_bf16(
          false, a0.v, false, b1.v, (short)0, acc[0][1], false, false);
      acc[1][1] = __builtin_amdgcn_wmma_f32_16x16x32_bf16(
          false, a1.v, false, b1.v, (short)0, acc[1][1], false, false);
      acc[0][2] = __builtin_amdgcn_wmma_f32_16x16x32_bf16(
          false, a0.v, false, b2.v, (short)0, acc[0][2], false, false);
      acc[1][2] = __builtin_amdgcn_wmma_f32_16x16x32_bf16(
          false, a1.v, false, b2.v, (short)0, acc[1][2], false, false);
      acc[0][3] = __builtin_amdgcn_wmma_f32_16x16x32_bf16(
          false, a0.v, false, b3.v, (short)0, acc[0][3], false, false);
      acc[1][3] = __builtin_amdgcn_wmma_f32_16x16x32_bf16(
          false, a1.v, false, b3.v, (short)0, acc[1][3], false, false);
    }
  }

  // D layout: lane -> column n = lane&15; VGPR r -> row r (+8 for hi half)
#pragma unroll
  for (int sub = 0; sub < 2; ++sub) {
    const size_t obase = (size_t)(blockIdx.x * 32 + sub * 16) * Cout +
                         (size_t)blockIdx.y * 64 + m;
#pragma unroll
    for (int q = 0; q < 4; ++q) {
      float* ocol = out + obase + q * 16;
#pragma unroll
      for (int r2 = 0; r2 < 8; ++r2) {
        ocol[(size_t)(r2 + hi * 8) * Cout] = acc[sub][q][r2];
      }
    }
  }
}

// ---------------------------------------------------------------------------
// Fused GroupNorm (group size 16 channels, per voxel) + optional residual +
// ReLU + f32 -> bf16, writing with a channel offset/stride (free concat).
// ---------------------------------------------------------------------------
__global__ void gn_relu_bf16(const float* __restrict__ src,
                             const unsigned short* __restrict__ resid,
                             const float* __restrict__ gamma,
                             const float* __restrict__ beta,
                             unsigned short* __restrict__ dst,
                             int V, int C, int dstStride, int dstOff)
{
  const int G = C >> 4;
  const int idx = blockIdx.x * blockDim.x + threadIdx.x;
  if (idx >= V * G) return;
  const int v = idx / G;
  const int g = idx - v * G;

  const float* s = src + (size_t)v * C + g * 16;
  float x[16], mu = 0.f;
#pragma unroll
  for (int j = 0; j < 16; ++j) { x[j] = s[j]; mu += x[j]; }
  mu *= 0.0625f;
  float var = 0.f;
#pragma unroll
  for (int j = 0; j < 16; ++j) { const float d = x[j] - mu; var += d * d; }
  const float inv = rsqrtf(var * 0.0625f + 1e-5f);

  unsigned short* o = dst + (size_t)v * dstStride + dstOff + g * 16;
#pragma unroll
  for (int j = 0; j < 16; ++j) {
    float y = (x[j] - mu) * inv * gamma[g * 16 + j] + beta[g * 16 + j];
    if (resid) y += bf2f(resid[(size_t)v * C + g * 16 + j]);
    y = fmaxf(y, 0.f);
    o[j] = f2bf(y);
  }
}

// ---------------------------------------------------------------------------
// Pack OIDHW f32 weights into WMMA bf16 B-fragment order, cout-tile innermost:
// block[(t*KC + kc)*NCT + ct] = 32 lanes x 16 halfs, contiguous per lane.
// ---------------------------------------------------------------------------
__global__ void pack_weights(const float* __restrict__ w,
                             unsigned short* __restrict__ wpk,
                             int Cin, int Cout, int T, int total)
{
  const int idx = blockIdx.x * blockDim.x + threadIdx.x;
  if (idx >= total) return;
  const int e = idx & 15;
  const int l = (idx >> 4) & 31;
  int rest = idx >> 9;
  const int NCT = Cout >> 4;
  const int KC  = Cin >> 5;
  const int ct = rest % NCT; rest /= NCT;
  const int kc = rest % KC;
  const int t  = rest / KC;

  const int n  = ct * 16 + (l & 15);                                // cout
  const int kk = (l < 16) ? (e < 8 ? e : e + 8) : (e < 8 ? e + 8 : e + 16);
  const int cin = kc * 32 + kk;
  const size_t src = ((size_t)n * Cin + cin) * T + t;               // OIDHW
  wpk[idx] = f2bf(w[src]);
}

// --------------------------- layout shufflers ------------------------------
__global__ void zero_u16(unsigned short* __restrict__ p, int n)
{
  const int idx = blockIdx.x * blockDim.x + threadIdx.x;
  if (idx < n) p[idx] = 0;
}

__global__ void nchw_to_cl_bf16(const float* __restrict__ x,
                                unsigned short* __restrict__ o, int V, int C)
{
  const int idx = blockIdx.x * blockDim.x + threadIdx.x;
  if (idx >= V * C) return;
  const int v = idx / C, c = idx - v * C;
  o[idx] = f2bf(x[(size_t)c * V + v]);
}

__global__ void copy_cl(const unsigned short* __restrict__ src,
                        unsigned short* __restrict__ dst,
                        int V, int C, int dstStride, int dstOff)
{
  const int idx = blockIdx.x * blockDim.x + threadIdx.x;
  if (idx >= V * C) return;
  const int v = idx / C, c = idx - v * C;
  dst[(size_t)v * dstStride + dstOff + c] = src[idx];
}

__global__ void cl_to_nchw_f32(const unsigned short* __restrict__ src,
                               float* __restrict__ dst, int V, int C)
{
  const int idx = blockIdx.x * blockDim.x + threadIdx.x;
  if (idx >= V * C) return;
  const int c = idx / V, v = idx - c * V;
  dst[idx] = bf2f(src[(size_t)v * C + c]);
}

// ---------------------------------------------------------------------------
extern "C" void kernel_launch(void* const* d_in, const int* in_sizes, int n_in,
                              void* d_out, int out_size, void* d_ws,
                              size_t ws_size, hipStream_t stream)
{
  (void)in_sizes; (void)n_in; (void)out_size; (void)ws_size;
  const int V0 = 48 * 48 * 48, V1 = 24 * 24 * 24, V2 = 12 * 12 * 12;

  size_t off = 0;
  auto alloc = [&](size_t bytes) -> void* {
    void* p = (char*)d_ws + off;
    off += (bytes + 255) & ~(size_t)255;
    return p;
  };

  unsigned short* zrow  = (unsigned short*)alloc(1024 * 2);             // zero row
  unsigned short* aIn   = (unsigned short*)alloc((size_t)V0 * 64  * 2); // x0 lives here
  unsigned short* aT0   = (unsigned short*)alloc((size_t)V0 * 64  * 2);
  unsigned short* aX1   = (unsigned short*)alloc((size_t)V1 * 128 * 2);
  unsigned short* aT1   = (unsigned short*)alloc((size_t)V1 * 128 * 2);
  unsigned short* aY1   = (unsigned short*)alloc((size_t)V1 * 128 * 2);
  unsigned short* aX2   = (unsigned short*)alloc((size_t)V2 * 128 * 2);
  unsigned short* aT2   = (unsigned short*)alloc((size_t)V2 * 128 * 2);
  unsigned short* aCat1 = (unsigned short*)alloc((size_t)V1 * 256 * 2);
  unsigned short* aCat0 = (unsigned short*)alloc((size_t)V0 * 128 * 2);
  unsigned short* aY0   = (unsigned short*)alloc((size_t)V0 * 64  * 2);
  float*          cv    = (float*)alloc((size_t)V0 * 64 * 4);           // conv f32 scratch

  zero_u16<<<4, 256, 0, stream>>>(zrow, 1024);

  // ------------------- pack all conv weights to bf16 fragments ------------
  struct WI { int idx, Cin, Cout, T; };
  const WI wtab[] = {
    {1, 64, 64, 27},  {4, 64, 64, 27},                    // res_down[0][0]
    {7, 128,128,27},  {10,128,128,27},                    // res_down[1][0]
    {13,128,128,27},  {16,128,128,27},                    // res_down[1][1]
    {19,128,128,27},  {22,128,128,27},                    // res_down[2][0]
    {25,128,128,27},  {28,128,128,27},                    // res_down[2][1]
    {31,128,128,27},  {34,128,128,27},                    // res_down[2][2]
    {37, 64,128,27},                                       // down[0]
    {40,128,128,27},                                       // down[1]
    {43,128,128,27},                                       // up[0]
    {46,128, 64,27},                                       // up[1]
    {49,256,128, 1},                                       // feat_adj[0]
    {52,128, 64, 1},                                       // feat_adj[1]
    {55,128,128,27},  {58,128,128,27},                    // res_up[0][0]
    {61,128,128,27},  {64,128,128,27},                    // res_up[0][1]
    {67, 64, 64,27},  {70, 64, 64,27},                    // res_up[1][0]
  };
  unsigned short* pk[80] = {};
  for (const WI& wi : wtab) {
    const int total = wi.Cin * wi.Cout * wi.T;
    pk[wi.idx] = (unsigned short*)alloc((size_t)total * 2);
    pack_weights<<<(total + 255) / 256, 256, 0, stream>>>(
        (const float*)d_in[wi.idx], pk[wi.idx], wi.Cin, wi.Cout, wi.T, total);
  }

  auto conv = [&](const unsigned short* act, const unsigned short* wp,
                  int Sin, int Sout, int Cin, int Cout, int mode) {
    const int Vout = Sout * Sout * Sout;
    dim3 g(Vout / 32, Cout / 64, 1);
    const int KC = Cin >> 5;
#define LCH(M, K) conv3d_wmma<M, K><<<g, 32, 0, stream>>>( \
        act, wp, zrow, cv, Sin, Sout, Cout)
    if      (mode == 0 && KC == 2) LCH(0, 2);
    else if (mode == 0)            LCH(0, 4);
    else if (mode == 1 && KC == 2) LCH(1, 2);
    else if (mode == 1)            LCH(1, 4);
    else if (mode == 2)            LCH(2, 4);
    else if (KC == 8)              LCH(3, 8);
    else                           LCH(3, 4);
#undef LCH
  };
  auto gn = [&](const unsigned short* resid, int gwi, int gbi,
                unsigned short* dst, int V, int C, int stride, int offc) {
    const int n = V * (C >> 4);
    gn_relu_bf16<<<(n + 255) / 256, 256, 0, stream>>>(
        cv, resid, (const float*)d_in[gwi], (const float*)d_in[gbi],
        dst, V, C, stride, offc);
  };
  auto resblock = [&](unsigned short* a, unsigned short* tmp, int base,
                      int S, int C) {
    const int V = S * S * S;
    conv(a, pk[base], S, S, C, C, 0);
    gn(nullptr, base + 1, base + 2, tmp, V, C, C, 0);
    conv(tmp, pk[base + 3], S, S, C, C, 0);
    gn(a, base + 4, base + 5, a, V, C, C, 0);   // + residual, ReLU, in-place
  };

  // ------------------------------ forward ---------------------------------
  // input NCDHW f32 -> channels-last bf16
  nchw_to_cl_bf16<<<(V0 * 64 + 255) / 256, 256, 0, stream>>>(
      (const float*)d_in[0], aIn, V0, 64);

  // down path
  resblock(aIn, aT0, 1, 48, 64);                              // x0 = aIn
  conv(aIn, pk[37], 48, 24, 64, 128, 1);                      // down[0]
  gn(nullptr, 38, 39, aX1, V1, 128, 128, 0);
  resblock(aX1, aT1, 7, 24, 128);
  resblock(aX1, aT1, 13, 24, 128);                            // x1 = aX1
  conv(aX1, pk[40], 24, 12, 128, 128, 1);                     // down[1]
  gn(nullptr, 41, 42, aX2, V2, 128, 128, 0);
  resblock(aX2, aT2, 19, 12, 128);
  resblock(aX2, aT2, 25, 12, 128);
  resblock(aX2, aT2, 31, 12, 128);                            // x2 = aX2

  // up path, level 1
  conv(aX2, pk[43], 12, 24, 128, 128, 2);                     // up[0] (transpose)
  gn(nullptr, 44, 45, aCat1, V1, 128, 256, 0);                // u -> cat[:,0:128]
  copy_cl<<<(V1 * 128 + 255) / 256, 256, 0, stream>>>(
      aX1, aCat1, V1, 128, 256, 128);                         // x1 -> cat[:,128:256]
  conv(aCat1, pk[49], 24, 24, 256, 128, 3);                   // feat_adj[0] 1x1
  gn(nullptr, 50, 51, aY1, V1, 128, 128, 0);
  resblock(aY1, aT1, 55, 24, 128);
  resblock(aY1, aT1, 61, 24, 128);                            // y1 = aY1

  // up path, level 0
  conv(aY1, pk[46], 24, 48, 128, 64, 2);                      // up[1] (transpose)
  gn(nullptr, 47, 48, aCat0, V0, 64, 128, 0);                 // u -> cat[:,0:64]
  copy_cl<<<(V0 * 64 + 255) / 256, 256, 0, stream>>>(
      aIn, aCat0, V0, 64, 128, 64);                           // x0 -> cat[:,64:128]
  conv(aCat0, pk[52], 48, 48, 128, 64, 3);                    // feat_adj[1] 1x1
  gn(nullptr, 53, 54, aY0, V0, 64, 64, 0);
  resblock(aY0, aT0, 67, 48, 64);                             // y0 = aY0

  // ------------------------------ outputs ---------------------------------
  float* dout = (float*)d_out;
  cl_to_nchw_f32<<<(V2 * 128 + 255) / 256, 256, 0, stream>>>(aX2, dout, V2, 128);
  cl_to_nchw_f32<<<(V1 * 128 + 255) / 256, 256, 0, stream>>>(
      aY1, dout + (size_t)128 * V2, V1, 128);
  cl_to_nchw_f32<<<(V0 * 64 + 255) / 256, 256, 0, stream>>>(
      aY0, dout + (size_t)128 * V2 + (size_t)128 * V1, V0, 64);
}
